// LowRankRNN_8495445311624
// MI455X (gfx1250) — matrix operand order repaired and checked by reading, not compiled, not run on hardware
//
#include <hip/hip_runtime.h>

#define ALPHA 0.1f
#define B_DIM 64
#define T_DIM 512
#define D_IN  128
#define H_DIM 1024
#define R_DIM 8

typedef __attribute__((ext_vector_type(16))) _Float16 v16h;
typedef __attribute__((ext_vector_type(8)))  _Float16 half8;
typedef __attribute__((ext_vector_type(4)))  _Float16 h4;
typedef __attribute__((ext_vector_type(8)))  float    v8f;
typedef __attribute__((ext_vector_type(4)))  float    f4;

// ---------------------------------------------------------------------------
// Kernel 1: u = x @ I^T   (M=B*T=32768, K=128, N=H=1024), fp32 in/out,
// f16 WMMA compute with fp32 accumulation. Result written into d_out.
// WG = 256 threads (8 waves). Tile: 128(M) x 64(N), full K=128 staged in LDS.
// ---------------------------------------------------------------------------
#define LDA 136   // 128 + 8 halves pad (keeps 16B alignment, breaks bank stride)
#define LDB 136

__device__ __forceinline__ h4 cvt4(f4 v) {
    h4 r;
    r.x = (_Float16)v.x; r.y = (_Float16)v.y;
    r.z = (_Float16)v.z; r.w = (_Float16)v.w;
    return r;
}

__global__ __launch_bounds__(256)
void k_input_gemm(const float* __restrict__ x, const float* __restrict__ I,
                  float* __restrict__ u)
{
    __shared__ _Float16 As[128 * LDA];   // [M=128][K=128] row-major (K contiguous)
    __shared__ _Float16 Bs[64  * LDB];   // [N=64 ][K=128] row-major (K contiguous)

    const int tid  = threadIdx.x;
    const int lane = tid & 31;
    const int wave = tid >> 5;
    const int m0   = blockIdx.y * 128;
    const int n0   = blockIdx.x * 64;

    // --- stage A: x rows m0..m0+127, 128 f32 each -> f16 LDS (4096 float4) ---
    {
        const f4* xg = (const f4*)(x + (size_t)m0 * D_IN);
        #pragma unroll
        for (int it = 0; it < 16; ++it) {
            int e   = it * 256 + tid;      // 0..4095
            int row = e >> 5;              // 32 float4 per row
            int c4  = e & 31;
            *(h4*)&As[row * LDA + c4 * 4] = cvt4(xg[e]);
        }
    }
    // --- stage B: I rows n0..n0+63, 128 f32 each -> f16 LDS (2048 float4) ---
    {
        const f4* ig = (const f4*)(I + (size_t)n0 * D_IN);
        #pragma unroll
        for (int it = 0; it < 8; ++it) {
            int e   = it * 256 + tid;      // 0..2047
            int row = e >> 5;
            int c4  = e & 31;
            *(h4*)&Bs[row * LDB + c4 * 4] = cvt4(ig[e]);
        }
    }
    __syncthreads();

    // Each wave: 16 M-rows x 64 N-cols = 4 accumulator tiles, K=128 (4 wmma each)
    v8f acc[4] = {v8f{}, v8f{}, v8f{}, v8f{}};
    const int msub  = wave * 16;
    const int arow  = msub + (lane & 15);
    const int ahalf = ((lane >> 4) & 1) * 8;    // A: K-offset of first 8 halves
    const int bhalf = ((lane >> 4) & 1) * 8;    // B: K-offset of first 8 halves

    #pragma unroll
    for (int kk = 0; kk < 4; ++kk) {
        const int K0 = kk * 32;
        // A fragment: halves[0..7] = K0+ahalf.., halves[8..15] = K0+16+ahalf..
        half8 alo = *(const half8*)&As[arow * LDA + K0 + ahalf];
        half8 ahi = *(const half8*)&As[arow * LDA + K0 + ahalf + 16];
        v16h a;
        #pragma unroll
        for (int i = 0; i < 8; ++i) { a[i] = alo[i]; a[i + 8] = ahi[i]; }

        #pragma unroll
        for (int nt = 0; nt < 4; ++nt) {
            const int brow = nt * 16 + (lane & 15);
            half8 blo = *(const half8*)&Bs[brow * LDB + K0 + bhalf];
            half8 bhi = *(const half8*)&Bs[brow * LDB + K0 + bhalf + 16];
            v16h b;
            #pragma unroll
            for (int i = 0; i < 8; ++i) { b[i] = blo[i]; b[i + 8] = bhi[i]; }

            acc[nt] = __builtin_amdgcn_wmma_f32_16x16x32_f16(
                false, a, false, b, (short)0, acc[nt], false, false);
        }
    }

    // C layout: VGPR v, lane L -> row M = v + (L>=16 ? 8 : 0), col N = L&15
    const int rbase = m0 + msub + ((lane >> 4) << 3);
    #pragma unroll
    for (int nt = 0; nt < 4; ++nt) {
        const int col = n0 + nt * 16 + (lane & 15);
        #pragma unroll
        for (int v = 0; v < 8; ++v) {
            u[(size_t)(rbase + v) * H_DIM + col] = acc[nt][v];
        }
    }
}

// ---------------------------------------------------------------------------
// Kernel 2: sequential scan. One WG per batch (64 WGs), 256 threads (8 waves),
// each thread owns 4 contiguous h plus its rows of m and n in registers.
//   h <- (1-a)h + a*((tanh(h)@n)@m^T + u_t) ; out[b,t,:] = h   (in place over u)
// One barrier per step via double-buffered LDS partials.
// ---------------------------------------------------------------------------
__global__ __launch_bounds__(256)
void k_scan(const float* __restrict__ m, const float* __restrict__ n,
            float* __restrict__ out /* [B][T][H]: holds u, overwritten with h */)
{
    __shared__ __align__(16) float parts[2][8][8];  // [buf][wave][r]

    const int b    = blockIdx.x;
    const int tid  = threadIdx.x;
    const int lane = tid & 31;
    const int wave = tid >> 5;
    const int h0   = tid * 4;

    // Preload this thread's weight slices: m,n are [H][R=8] row-major.
    float mw[4][8], nw[4][8];
    #pragma unroll
    for (int i = 0; i < 4; ++i) {
        f4 a = *(const f4*)(m + (size_t)(h0 + i) * R_DIM);
        f4 c = *(const f4*)(m + (size_t)(h0 + i) * R_DIM + 4);
        mw[i][0]=a.x; mw[i][1]=a.y; mw[i][2]=a.z; mw[i][3]=a.w;
        mw[i][4]=c.x; mw[i][5]=c.y; mw[i][6]=c.z; mw[i][7]=c.w;
        f4 d = *(const f4*)(n + (size_t)(h0 + i) * R_DIM);
        f4 e = *(const f4*)(n + (size_t)(h0 + i) * R_DIM + 4);
        nw[i][0]=d.x; nw[i][1]=d.y; nw[i][2]=d.z; nw[i][3]=d.w;
        nw[i][4]=e.x; nw[i][5]=e.y; nw[i][6]=e.z; nw[i][7]=e.w;
    }

    float h[4] = {0.f, 0.f, 0.f, 0.f};
    float* outb = out + (size_t)b * T_DIM * H_DIM;

    for (int t = 0; t < T_DIM; ++t) {
        float* rowp = outb + (size_t)t * H_DIM + h0;
        f4 uvec = *(const f4*)rowp;          // u_t (issued early; waited late)

        float th[4];
        #pragma unroll
        for (int i = 0; i < 4; ++i) th[i] = tanhf(h[i]);

        // rank-R partials over this thread's 4 elements
        float p[8];
        #pragma unroll
        for (int r = 0; r < 8; ++r)
            p[r] = th[0]*nw[0][r] + th[1]*nw[1][r] + th[2]*nw[2][r] + th[3]*nw[3][r];

        // wave32 butterfly reduction (all lanes end with the wave sum)
        #pragma unroll
        for (int s = 1; s < 32; s <<= 1) {
            #pragma unroll
            for (int r = 0; r < 8; ++r) p[r] += __shfl_xor(p[r], s, 32);
        }

        const int buf = t & 1;
        if (lane == 0) {
            f4 p0 = {p[0], p[1], p[2], p[3]};
            f4 p1 = {p[4], p[5], p[6], p[7]};
            *(f4*)&parts[buf][wave][0] = p0;
            *(f4*)&parts[buf][wave][4] = p1;
        }
        __syncthreads();

        // sum the 8 wave partials (broadcast LDS reads, vectorized)
        f4 s0 = {0.f, 0.f, 0.f, 0.f}, s1 = {0.f, 0.f, 0.f, 0.f};
        #pragma unroll
        for (int w = 0; w < 8; ++w) {
            s0 += *(const f4*)&parts[buf][w][0];
            s1 += *(const f4*)&parts[buf][w][4];
        }
        float v[8] = {s0.x, s0.y, s0.z, s0.w, s1.x, s1.y, s1.z, s1.w};

        float uu[4] = {uvec.x, uvec.y, uvec.z, uvec.w};
        #pragma unroll
        for (int i = 0; i < 4; ++i) {
            float rh = 0.f;
            #pragma unroll
            for (int r = 0; r < 8; ++r) rh += v[r] * mw[i][r];
            h[i] = (1.0f - ALPHA) * h[i] + ALPHA * (rh + uu[i]);
        }
        f4 hv = {h[0], h[1], h[2], h[3]};
        *(f4*)rowp = hv;                      // overwrite u_t with h_t
    }
}

// ---------------------------------------------------------------------------
extern "C" void kernel_launch(void* const* d_in, const int* in_sizes, int n_in,
                              void* d_out, int out_size, void* d_ws, size_t ws_size,
                              hipStream_t stream) {
    const float* x = (const float*)d_in[0];   // [B,T,D_IN]
    const float* m = (const float*)d_in[1];   // [H,R]
    const float* n = (const float*)d_in[2];   // [H,R]
    const float* I = (const float*)d_in[3];   // [H,D_IN]
    float* out = (float*)d_out;               // [B,T,H]

    // Kernel 1: u = x @ I^T into d_out. Grid covers M=32768 x N=1024 fully.
    dim3 g1(H_DIM / 64, (B_DIM * T_DIM) / 128);
    k_input_gemm<<<g1, 256, 0, stream>>>(x, I, out);

    // Kernel 2: in-place sequential scan over T (stream-ordered after k1).
    k_scan<<<dim3(B_DIM), 256, 0, stream>>>(m, n, out);
}
